// Attention_57801669869899
// MI455X (gfx1250) — matrix operand order repaired and checked by reading, not compiled, not run on hardware
//
#include <hip/hip_runtime.h>
#include <hip/hip_bf16.h>

#define T_TOK 2048
#define NX_   512
#define HD_   128
#define LLM   64
#define KNB   32
#define BB    2

#define BM 256   // block M tile (8 waves x 32 rows)
#define BN 64    // block N tile
#define BK 32    // K step

typedef __attribute__((ext_vector_type(16))) __bf16 v16bf;
typedef __attribute__((ext_vector_type(8)))  __bf16 v8bf;
typedef __attribute__((ext_vector_type(4)))  __bf16 v4bf;
typedef __attribute__((ext_vector_type(8)))  float  v8f;

__device__ __forceinline__ v16bf cat16(v8bf lo, v8bf hi) {
  v16bf r;
#pragma unroll
  for (int e = 0; e < 8; ++e) { r[e] = lo[e]; r[8 + e] = hi[e]; }
  return r;
}

// ---- CDNA5 async global->LDS copy (ASYNCcnt-tracked DMA, no VGPR data) ----
__device__ __forceinline__ unsigned lds_off(const void* p) {
  return (unsigned)(size_t)p;  // low 32 bits of flat LDS address = LDS offset
}
__device__ __forceinline__ void async_b128(unsigned lds_addr, const __bf16* g) {
  asm volatile("global_load_async_to_lds_b128 %0, %1, off"
               :: "v"(lds_addr), "v"((unsigned long long)(size_t)g)
               : "memory");
}
__device__ __forceinline__ void wait_async0() {
  asm volatile("s_wait_asynccnt 0x0" ::: "memory");
}

// Cooperative load of one K-stage: A tile BMxBK, B tile BNxBK (bf16).
__device__ __forceinline__ void coop_load_tiles(
    const __bf16* __restrict__ A, int lda, const __bf16* __restrict__ W, int ldw,
    int blkM, int n0blk, int k0, __bf16* sA, __bf16* sB, int stage) {
  int tid = threadIdx.x;
  // A tile: 256 rows x 32 cols; 1 thread/row, 64B = 4 x b128
  const __bf16* gA = A + (size_t)(blkM + tid) * lda + k0;
  __bf16* lA = sA + stage * (BM * BK) + tid * BK;
  unsigned la = lds_off(lA);
  async_b128(la,      gA);
  async_b128(la + 16, gA + 8);
  async_b128(la + 32, gA + 16);
  async_b128(la + 48, gA + 24);
  // B tile: 64 rows x 32 cols; 4 threads/row, 16B each (1 x b128)
  int rowB = tid >> 2;
  int colB = (tid & 3) << 3;
  const __bf16* gB = W + (size_t)(n0blk + rowB) * ldw + k0 + colB;
  __bf16* lB = sB + stage * (BN * BK) + rowB * BK + colB;
  async_b128(lds_off(lB), gB);
}

// Per-wave compute on one K-stage: 32x64 tile = 8 WMMAs.
// All fragment ds_loads are issued first, then WMMAs go back-to-back.
__device__ __forceinline__ void compute_stage(const __bf16* sA, const __bf16* sB,
                                              int stage, int wave, int lane,
                                              v8f acc[2][4]) {
  int ml   = lane & 15;
  int kb8  = (lane >> 4) << 3;   // A frag K sub-offset
  int kb16 = (lane >> 4) << 4;   // unused (B frag uses kb8, same as A)
  const __bf16* pa0 = sA + stage * (BM * BK) + (wave * 32 + ml) * BK + kb8;
  const __bf16* pa1 = pa0 + 16 * BK;
  v16bf a0 = cat16(*(const v8bf*)pa0, *(const v8bf*)(pa0 + 16));
  v16bf a1 = cat16(*(const v8bf*)pa1, *(const v8bf*)(pa1 + 16));
  v16bf bb[4];
#pragma unroll
  for (int j = 0; j < 4; ++j) {
    const __bf16* pb = sB + stage * (BN * BK) + (j * 16 + ml) * BK + kb8;
    bb[j] = cat16(*(const v8bf*)pb, *(const v8bf*)(pb + 16));
  }
#pragma unroll
  for (int j = 0; j < 4; ++j)
    acc[0][j] = __builtin_amdgcn_wmma_f32_16x16x32_bf16(
        false, a0, false, bb[j], (short)0, acc[0][j], false, false);
#pragma unroll
  for (int j = 0; j < 4; ++j)
    acc[1][j] = __builtin_amdgcn_wmma_f32_16x16x32_bf16(
        false, a1, false, bb[j], (short)0, acc[1][j], false, false);
}

// Double-buffered, async-pipelined block GEMM: C = A @ W^T (bf16 in, f32 acc).
__device__ __forceinline__ void gemm_block(const __bf16* __restrict__ A, int lda,
                                           const __bf16* __restrict__ W, int ldw,
                                           float* __restrict__ Cf, __bf16* __restrict__ Cb,
                                           int ldc, int K, int blkM, int n0blk,
                                           __bf16* sA, __bf16* sB) {
  int lane = threadIdx.x & 31;
  int wave = threadIdx.x >> 5;
  v8f z = {0.f, 0.f, 0.f, 0.f, 0.f, 0.f, 0.f, 0.f};
  v8f acc[2][4] = {{z, z, z, z}, {z, z, z, z}};

  coop_load_tiles(A, lda, W, ldw, blkM, n0blk, 0, sA, sB, 0);
  wait_async0();
  __syncthreads();
  int s = 0;
  for (int k0 = 0; k0 < K; k0 += BK) {
    int nk = k0 + BK;
    if (nk < K) coop_load_tiles(A, lda, W, ldw, blkM, n0blk, nk, sA, sB, s ^ 1);
    compute_stage(sA, sB, s, wave, lane, acc);  // overlaps with async DMA
    if (nk < K) {
      wait_async0();
      __syncthreads();
    }
    s ^= 1;
  }
  // D layout: VGPR r <-> M = r + 8*(lane>=16), N = lane&15
  int nlo = lane & 15;
  int mb = blkM + wave * 32 + ((lane >> 4) << 3);
#pragma unroll
  for (int i = 0; i < 2; ++i) {
#pragma unroll
    for (int j = 0; j < 4; ++j) {
      int n = n0blk + j * 16 + nlo;
      int m0 = mb + i * 16;
      if (Cf) {
#pragma unroll
        for (int r = 0; r < 8; ++r) Cf[(size_t)(m0 + r) * ldc + n] = acc[i][j][r];
      }
      if (Cb) {
#pragma unroll
        for (int r = 0; r < 8; ++r) Cb[(size_t)(m0 + r) * ldc + n] = (__bf16)acc[i][j][r];
      }
    }
  }
}

// ---------------- kernels ----------------

__global__ void cvt_kernel(const float* __restrict__ src, __bf16* __restrict__ dst, int n) {
  int i = (blockIdx.x * blockDim.x + threadIdx.x) * 4;
  if (i < n) {
    const float4 f = *(const float4*)(src + i);
    v4bf o = {(__bf16)f.x, (__bf16)f.y, (__bf16)f.z, (__bf16)f.w};
    *(v4bf*)(dst + i) = o;
  }
}

// Grid: (M/256, N/64). Block: 256 threads = 8 waves stacked along M.
__global__ __launch_bounds__(256) void gemm_bf16_kernel(
    const __bf16* __restrict__ A, int lda, const __bf16* __restrict__ W, int ldw,
    float* __restrict__ Cf, __bf16* __restrict__ Cb, int ldc, int K) {
  __shared__ __bf16 sA[2 * BM * BK];
  __shared__ __bf16 sB[2 * BN * BK];
  gemm_block(A, lda, W, ldw, Cf, Cb, ldc, K, blockIdx.x * BM, blockIdx.y * BN, sA, sB);
}

// w1 = q_rows @ skT  and  w2T = k_rows @ sq, both [T,64] f32 per (b,h).
// Grid: (T/256, 1, 2*B*4). Block 256.
__global__ __launch_bounds__(256) void w12_kernel(
    const __bf16* __restrict__ qb, const __bf16* __restrict__ kb,
    const __bf16* __restrict__ sqb, const __bf16* __restrict__ skTb,
    float* __restrict__ w1, float* __restrict__ w2T) {
  __shared__ __bf16 sA[2 * BM * BK];
  __shared__ __bf16 sB[2 * BN * BK];
  int z = blockIdx.z;
  int which = z >> 3;  // 0 -> w1, 1 -> w2T
  int b = (z >> 2) & 1;
  int h = z & 3;
  const __bf16* A  = (which == 0 ? qb : kb) + (size_t)b * T_TOK * NX_ + h * HD_;
  const __bf16* Wm = (which == 0 ? skTb : sqb) + (size_t)((b * 4 + h) * LLM) * HD_;
  float* C = (which == 0 ? w1 : w2T) + (size_t)(b * 4 + h) * T_TOK * LLM;
  gemm_block(A, NX_, Wm, HD_, C, (__bf16*)nullptr, LLM, HD_, blockIdx.x * BM, 0, sA, sB);
}

// Grid: (L, 4, B). Block 128. Normalize landmark q/k vectors, scale by m1/m2.
__global__ __launch_bounds__(128) void landmark_kernel(
    const __bf16* __restrict__ qb, const __bf16* __restrict__ kb,
    const float* __restrict__ m1, const float* __restrict__ m2,
    const int* __restrict__ lmk, __bf16* __restrict__ sqb, __bf16* __restrict__ skTb) {
  int i = blockIdx.x, h = blockIdx.y, b = blockIdx.z, d = threadIdx.x;
  __shared__ float rq[128], rk[128];
  int t = lmk[i];
  size_t src = (size_t)(b * T_TOK + t) * NX_ + h * HD_ + d;
  float qv = (float)qb[src];
  float kv = (float)kb[src];
  rq[d] = qv * qv; rk[d] = kv * kv;
  __syncthreads();
  for (int s = 64; s > 0; s >>= 1) {
    if (d < s) { rq[d] += rq[d + s]; rk[d] += rk[d + s]; }
    __syncthreads();
  }
  float sq_scale = m1[h * LLM + i] * rsqrtf(rq[0] + 1e-20f);
  float sk_scale = m2[h * LLM + i] * rsqrtf(rk[0] + 1e-20f);
  size_t o = (size_t)((b * 4 + h) * LLM + i) * HD_ + d;
  sqb[o]  = (__bf16)(qv * sq_scale);
  skTb[o] = (__bf16)(kv * sk_scale);
}

// Grid: (B, NX/128). Block 128. Column means of v per batch.
__global__ __launch_bounds__(128) void vmean_kernel(const float* __restrict__ vf,
                                                    float* __restrict__ vmean) {
  int b = blockIdx.x;
  int d = blockIdx.y * 128 + threadIdx.x;
  float acc = 0.f;
  for (int t = 0; t < T_TOK; ++t) acc += vf[(size_t)(b * T_TOK + t) * NX_ + d];
  vmean[b * NX_ + d] = acc * (1.0f / T_TOK);
}

__global__ __launch_bounds__(128) void lam_kernel(
    const float* __restrict__ lq1, const float* __restrict__ lk1,
    const float* __restrict__ lq2, const float* __restrict__ lk2,
    const int* __restrict__ lval, float* __restrict__ lamBuf) {
  __shared__ float r1[128], r2[128];
  int tid = threadIdx.x;
  r1[tid] = lq1[tid] * lk1[tid];
  r2[tid] = lq2[tid] * lk2[tid];
  __syncthreads();
  for (int s = 64; s > 0; s >>= 1) {
    if (tid < s) { r1[tid] += r1[tid + s]; r2[tid] += r2[tid + s]; }
    __syncthreads();
  }
  if (tid == 0) {
    float lam_init = 0.8f - 0.6f * __expf(-0.3f * (float)lval[0]);
    lamBuf[0] = __expf(r1[0]) - __expf(r2[0]) + lam_init;  // lam
    lamBuf[1] = lam_init;
  }
}

// One block per (b,t). Sparse masked softmax-attention + RMS norm, bf16 out.
__global__ __launch_bounds__(256) void sparse_attn_kernel(
    const float* __restrict__ w1, const float* __restrict__ w2T,
    const float* __restrict__ vf, const float* __restrict__ vmean,
    const int* __restrict__ rns, const float* __restrict__ rmsg,
    const float* __restrict__ lamBuf, __bf16* __restrict__ abf) {
  int b = blockIdx.x >> 11;
  int t = blockIdx.x & (T_TOK - 1);
  int tid = threadIdx.x;
  __shared__ int   sIdx[KNB], sFlag[KNB], sS[KNB], sCnt;
  __shared__ float sW1[4][LLM];
  __shared__ float sWv[4][KNB];
  __shared__ float sMix[2][KNB];
  __shared__ float sSumSq[2];

  const int* rowIdx = rns + (size_t)(b * T_TOK + t) * KNB;
  if (tid < KNB) sIdx[tid] = rowIdx[tid];
  if (tid < 2) sSumSq[tid] = 0.f;
  __syncthreads();

  // Symmetric mask membership with de-dup (dense scatter collapses duplicates).
  if (tid < KNB) {
    int j = sIdx[tid];
    int ok = 1;
    for (int u = 0; u < tid; ++u)
      if (sIdx[u] == j) { ok = 0; break; }
    if (ok) {
      const int* jr = rns + (size_t)(b * T_TOK + j) * KNB;
      int mem = 0;
#pragma unroll
      for (int u = 0; u < KNB; ++u) mem |= (jr[u] == t);
      ok = mem;
    }
    sFlag[tid] = ok;
  }
  {  // stage the 4 w1 rows (4 x 64 f32) in LDS
    int h = tid >> 6, d = tid & 63;
    sW1[h][d] = w1[(size_t)((b * 4 + h) * T_TOK + t) * LLM + d];
  }
  __syncthreads();
  if (tid == 0) {
    int c = 0;
    for (int u = 0; u < KNB; ++u)
      if (sFlag[u]) sS[c++] = sIdx[u];
    sCnt = c;
  }
  __syncthreads();
  int cnt = sCnt;
  float lam = lamBuf[0], lam_init = lamBuf[1];

  if (cnt > 0) {
    if (tid < 4 * cnt) {  // w values: dot64(w1[h,t,:], w2T[h,j,:])
      int h = tid / cnt, s = tid - h * cnt;
      const float* w2row = w2T + (size_t)((b * 4 + h) * T_TOK + sS[s]) * LLM;
      float acc = 0.f;
#pragma unroll
      for (int d = 0; d < LLM; ++d) acc += sW1[h][d] * w2row[d];
      sWv[h][s] = acc;
    }
    __syncthreads();
    if (tid < 4) {  // per-head softmax over S (off-mask entries exp-underflow to 0)
      int h = tid;
      float mx = -3.0e38f;
      for (int s = 0; s < cnt; ++s) mx = fmaxf(mx, sWv[h][s]);
      float sum = 0.f;
      for (int s = 0; s < cnt; ++s) { float e = __expf(sWv[h][s] - mx); sWv[h][s] = e; sum += e; }
      float inv = 1.0f / sum;
      for (int s = 0; s < cnt; ++s) sWv[h][s] *= inv;
    }
    __syncthreads();
    if (tid < cnt) {  // wmix0 = (1-lam)p2 - lam p1 ; wmix1 = p2 - p1 + (1-2lam)p3
      float p1 = sWv[1][tid], p2 = sWv[2][tid], p3 = sWv[3][tid];
      sMix[0][tid] = (1.0f - lam) * p2 - lam * p1;
      sMix[1][tid] = p2 - p1 + (1.0f - 2.0f * lam) * p3;
    }
    __syncthreads();
  }

  // Each thread owns d0 = tid (v-head 0) and d1 = tid + 256 (v-head 1).
  int d0 = tid, d1 = tid + 256;
  float a0 = 0.f, a1 = 0.f;
  if (cnt > 0) {
    for (int s = 0; s < cnt; ++s) {
      const float* vr = vf + (size_t)(b * T_TOK + sS[s]) * NX_;
      a0 += sMix[0][s] * vr[d0];
      a1 += sMix[1][s] * vr[d1];
    }
  } else {  // fully-masked row: uniform softmax -> (1-2lam) * mean(v)
    float c = 1.0f - 2.0f * lam;
    a0 = c * vmean[b * NX_ + d0];
    a1 = c * vmean[b * NX_ + d1];
  }
  atomicAdd(&sSumSq[0], a0 * a0);
  atomicAdd(&sSumSq[1], a1 * a1);
  __syncthreads();
  float sc0 = rsqrtf(sSumSq[0] * (1.0f / 256.0f) + 1e-5f);
  float sc1 = rsqrtf(sSumSq[1] * (1.0f / 256.0f) + 1e-5f);
  float o0 = a0 * sc0 * rmsg[d0] * (1.0f - lam_init);
  float o1 = a1 * sc1 * rmsg[d1 - 256] * (1.0f - lam_init);
  size_t base = (size_t)(b * T_TOK + t) * NX_;
  abf[base + d0] = (__bf16)o0;
  abf[base + d1] = (__bf16)o1;
}

// ---------------- host ----------------

extern "C" void kernel_launch(void* const* d_in, const int* in_sizes, int n_in,
                              void* d_out, int out_size, void* d_ws, size_t ws_size,
                              hipStream_t stream) {
  (void)in_sizes; (void)n_in; (void)out_size; (void)ws_size;
  const float* x    = (const float*)d_in[0];
  const float* wq   = (const float*)d_in[1];
  const float* wk   = (const float*)d_in[2];
  const float* wvp  = (const float*)d_in[3];
  const float* wo   = (const float*)d_in[4];
  const float* m1   = (const float*)d_in[5];
  const float* m2   = (const float*)d_in[6];
  const float* lq1  = (const float*)d_in[7];
  const float* lk1  = (const float*)d_in[8];
  const float* lq2  = (const float*)d_in[9];
  const float* lk2  = (const float*)d_in[10];
  const float* rmsg = (const float*)d_in[11];
  const int*   lv   = (const int*)d_in[12];
  const int*   rns  = (const int*)d_in[14];
  const int*   lmk  = (const int*)d_in[15];
  float* out = (float*)d_out;

  char* p = (char*)d_ws;
  auto alloc = [&](size_t bytes) {
    void* r = (void*)p;
    p += (bytes + 255) & ~(size_t)255;
    return r;
  };
  const size_t MT = (size_t)BB * T_TOK;  // 4096 rows
  __bf16* xb   = (__bf16*)alloc(MT * NX_ * 2);
  __bf16* wqb  = (__bf16*)alloc((size_t)NX_ * NX_ * 2);
  __bf16* wkb  = (__bf16*)alloc((size_t)NX_ * NX_ * 2);
  __bf16* wvb  = (__bf16*)alloc((size_t)NX_ * NX_ * 2);
  __bf16* wob  = (__bf16*)alloc((size_t)NX_ * NX_ * 2);
  __bf16* qb   = (__bf16*)alloc(MT * NX_ * 2);
  __bf16* kb   = (__bf16*)alloc(MT * NX_ * 2);
  float*  vf   = (float*)alloc(MT * NX_ * 4);
  float*  vmn  = (float*)alloc((size_t)BB * NX_ * 4);
  __bf16* sqb  = (__bf16*)alloc((size_t)BB * 4 * LLM * HD_ * 2);
  __bf16* skTb = (__bf16*)alloc((size_t)BB * 4 * LLM * HD_ * 2);
  float*  w1b  = (float*)alloc((size_t)BB * 4 * T_TOK * LLM * 4);
  float*  w2Tb = (float*)alloc((size_t)BB * 4 * T_TOK * LLM * 4);
  __bf16* abf  = (__bf16*)alloc(MT * NX_ * 2);
  float*  lamB = (float*)alloc(256);

  const int nx2 = NX_ * NX_;
  cvt_kernel<<<(int)((MT * NX_) / 1024), 256, 0, stream>>>(x, xb, (int)(MT * NX_));
  cvt_kernel<<<nx2 / 1024, 256, 0, stream>>>(wq, wqb, nx2);
  cvt_kernel<<<nx2 / 1024, 256, 0, stream>>>(wk, wkb, nx2);
  cvt_kernel<<<nx2 / 1024, 256, 0, stream>>>(wvp, wvb, nx2);
  cvt_kernel<<<nx2 / 1024, 256, 0, stream>>>(wo, wob, nx2);

  dim3 gg((unsigned)(MT / BM), NX_ / BN, 1);
  gemm_bf16_kernel<<<gg, 256, 0, stream>>>(xb, NX_, wqb, NX_, nullptr, qb, NX_, NX_);
  gemm_bf16_kernel<<<gg, 256, 0, stream>>>(xb, NX_, wkb, NX_, nullptr, kb, NX_, NX_);
  gemm_bf16_kernel<<<gg, 256, 0, stream>>>(xb, NX_, wvb, NX_, vf, nullptr, NX_, NX_);

  landmark_kernel<<<dim3(LLM, 4, BB), 128, 0, stream>>>(qb, kb, m1, m2, lmk, sqb, skTb);
  w12_kernel<<<dim3(T_TOK / BM, 1, 2 * BB * 4), 256, 0, stream>>>(qb, kb, sqb, skTb, w1b, w2Tb);
  vmean_kernel<<<dim3(BB, NX_ / 128), 128, 0, stream>>>(vf, vmn);
  lam_kernel<<<1, 128, 0, stream>>>(lq1, lk1, lq2, lk2, lv, lamB);
  sparse_attn_kernel<<<BB * T_TOK, 256, 0, stream>>>(w1b, w2Tb, vf, vmn, rns, rmsg, lamB, abf);

  gemm_bf16_kernel<<<gg, 256, 0, stream>>>(abf, NX_, wob, NX_, out, nullptr, NX_, NX_);
}
